// FullDeepModelWithThreeStages_84413287236156
// MI455X (gfx1250) — matrix operand order repaired and checked by reading, not compiled, not run on hardware
//
#include <hip/hip_runtime.h>
#include <hip/hip_bf16.h>

typedef __attribute__((ext_vector_type(16))) _Float16 v16h;
typedef __attribute__((ext_vector_type(8)))  float    v8f;
typedef __attribute__((ext_vector_type(4)))  unsigned int u32x4;
typedef __attribute__((ext_vector_type(8)))  int      i32x8;
typedef __attribute__((ext_vector_type(4)))  int      i32x4;

#define DM      256
#define DFF     512
#define NHEAD   8
#define DKH     32
#define NLAYER  4
#define SEQ     2048
#define BATCH   2
#define ROWS    (BATCH*SEQ)
#define LN_EPS  1e-5f

// ---------------------------------------------------------------------------
// WMMA helpers (CDNA5 wave32, V_WMMA_F32_16X16X32_F16)
// ---------------------------------------------------------------------------
__device__ inline v8f wmma_f32_f16(v16h a, v16h b, v8f c) {
  return __builtin_amdgcn_wmma_f32_16x16x32_f16(
      /*neg_a=*/false, a, /*neg_b=*/false, b,
      /*c_mod=*/(short)0, c, /*reuse_a=*/false, /*reuse_b=*/false);
}

// A operand: 16(M) x 32(K) f16, row-major source with leading dim `ld`.
__device__ inline v16h load_frag_a(const _Float16* base, int ld) {
  const int lane = threadIdx.x & 31;
  const int r    = lane & 15;
  const int koff = (lane >> 4) << 4;           // 0 or 16
  const _Float16* p = base + (size_t)r * ld + koff;
  v16h out;
#pragma unroll
  for (int i = 0; i < 16; ++i) out[i]    = p[i];
  return out;
}

// B operand: 32(K) x 16(N) f16 read from an N-major buffer (contiguous in K).
__device__ inline v16h load_frag_b(const _Float16* base, int ld) {
  const int lane  = threadIdx.x & 31;
  const int c     = lane & 15;
  const int kbase = (lane >> 4) << 4;          // 0 or 16
  const _Float16* p = base + (size_t)c * ld + kbase;
  v16h out;
#pragma unroll
  for (int i = 0; i < 16; ++i) out[i] = p[i];
  return out;
}

// ---------------------------------------------------------------------------
// DPP butterfly reductions within each 16-lane half (no LDS traffic).
// quad_perm(xor1)=0xB1, quad_perm(xor2)=0x4E, row_half_mirror=0x141,
// row_mirror=0x140 together span lane bits 0..3.
// ---------------------------------------------------------------------------
__device__ inline float dpp_bcast_max16(float v) {
  union { float f; int i; } u, t;
  u.f = v;
  t.i = __builtin_amdgcn_update_dpp(0, u.i, 0xB1,  0xF, 0xF, true); u.f = fmaxf(u.f, t.f);
  t.i = __builtin_amdgcn_update_dpp(0, u.i, 0x4E,  0xF, 0xF, true); u.f = fmaxf(u.f, t.f);
  t.i = __builtin_amdgcn_update_dpp(0, u.i, 0x141, 0xF, 0xF, true); u.f = fmaxf(u.f, t.f);
  t.i = __builtin_amdgcn_update_dpp(0, u.i, 0x140, 0xF, 0xF, true); u.f = fmaxf(u.f, t.f);
  return u.f;
}
__device__ inline float dpp_bcast_sum16(float v) {
  union { float f; int i; } u, t;
  u.f = v;
  t.i = __builtin_amdgcn_update_dpp(0, u.i, 0xB1,  0xF, 0xF, true); u.f += t.f;
  t.i = __builtin_amdgcn_update_dpp(0, u.i, 0x4E,  0xF, 0xF, true); u.f += t.f;
  t.i = __builtin_amdgcn_update_dpp(0, u.i, 0x141, 0xF, 0xF, true); u.f += t.f;
  t.i = __builtin_amdgcn_update_dpp(0, u.i, 0x140, 0xF, 0xF, true); u.f += t.f;
  return u.f;
}

// ---------------------------------------------------------------------------
// Tensor Data Mover: async 2D f16 tile load global -> LDS (D# built inline).
// tile_w elems (dim0, contiguous), tile_h rows (dim1), row stride in elems.
// ---------------------------------------------------------------------------
__device__ inline void tdm_load_tile_f16(const _Float16* gptr, unsigned lds_off,
                                         unsigned tile_w, unsigned tile_h,
                                         unsigned stride_elems)
{
  const unsigned long long ga = (unsigned long long)gptr;
  u32x4 g0;
  g0[0] = 1u;                                            // count=1, user D#
  g0[1] = lds_off;                                       // lds_addr (bytes)
  g0[2] = (unsigned)(ga & 0xFFFFFFFFu);                  // global_addr[31:0]
  g0[3] = (unsigned)((ga >> 32) & 0x01FFFFFFu) | 0x80000000u; // [56:32] | type=2

  i32x8 g1;
  g1[0] = 0x00010000;                                    // data_size=1 (2B)
  g1[1] = (int)((tile_w & 0xFFFFu) << 16);               // tensor_dim0[15:0]
  g1[2] = (int)((tile_w >> 16) | ((tile_h & 0xFFFFu) << 16)); // td0 hi | td1 lo
  g1[3] = (int)((tile_h >> 16) | (tile_w << 16));        // td1 hi | tile_dim0
  g1[4] = (int)tile_h;                                   // tile_dim1 (tile_dim2=0)
  g1[5] = (int)stride_elems;                             // tensor_dim0_stride lo
  g1[6] = 0;                                             // stride hi | dim1_stride lo
  g1[7] = 0;                                             // dim1_stride hi

  const i32x4 gz = {0, 0, 0, 0};
#if defined(__clang_major__) && (__clang_major__ >= 23)
  const i32x8 gz8 = {0, 0, 0, 0, 0, 0, 0, 0};
  __builtin_amdgcn_tensor_load_to_lds(g0, g1, gz, gz, gz8, 0);
#else
  __builtin_amdgcn_tensor_load_to_lds(g0, g1, gz, gz, 0);
#endif
}

__device__ inline unsigned lds_offset_of(const void* p) {
  return (unsigned)(unsigned long long)(p);              // LDS aperture: addr[31:0]
}

// ---------------------------------------------------------------------------
// Tiled GEMM: C[M,N] = A[M,K](f16) * BT[N,K](f16)^T + bias.
// 64x64 block tile, 8 waves x (16x32) subtiles. Tiles staged by the TDM,
// double-buffered, synchronized with s_wait_tensorcnt.
// ---------------------------------------------------------------------------
template<bool RELU, bool OUT32, bool OUT16, bool OUT16T>
__global__ __launch_bounds__(256)
void gemm_f16_wmma(const _Float16* __restrict__ A,
                   const _Float16* __restrict__ BT,
                   const float*    __restrict__ bias,
                   float*          __restrict__ C32,
                   _Float16*       __restrict__ C16,
                   _Float16*       __restrict__ C16T,
                   int M, int N, int K)
{
  __shared__ __align__(16) _Float16 sA[2][64 * 32];
  __shared__ __align__(16) _Float16 sB[2][64 * 32];

  const int tid   = threadIdx.x;
  const int lane  = tid & 31;
  const int wid   = tid >> 5;
  const int m_off = (wid >> 1) << 4;     // 0,16,32,48
  const int n_off = (wid & 1) << 5;      // 0,32
  const int mBase = blockIdx.x * 64;
  const int nBase = blockIdx.y * 64;
  const int nk    = K >> 5;

  v8f c0 = {0.f,0.f,0.f,0.f,0.f,0.f,0.f,0.f};
  v8f c1 = {0.f,0.f,0.f,0.f,0.f,0.f,0.f,0.f};

  // prologue: wave 0 issues TDM for tiles 0 and 1
  if (tid < 32) {
    tdm_load_tile_f16(A  + (size_t)mBase * K, lds_offset_of(&sA[0][0]), 32, 64, K);
    tdm_load_tile_f16(BT + (size_t)nBase * K, lds_offset_of(&sB[0][0]), 32, 64, K);
    if (nk > 1) {
      tdm_load_tile_f16(A  + (size_t)mBase * K + 32, lds_offset_of(&sA[1][0]), 32, 64, K);
      tdm_load_tile_f16(BT + (size_t)nBase * K + 32, lds_offset_of(&sB[1][0]), 32, 64, K);
    }
  }

  for (int kt = 0; kt < nk; ++kt) {
    const int buf = kt & 1;
    if (tid < 32) {
      if (kt + 1 < nk) __builtin_amdgcn_s_wait_tensorcnt(2);  // tile kt landed
      else             __builtin_amdgcn_s_wait_tensorcnt(0);
    }
    __syncthreads();

    v16h af  = load_frag_a(&sA[buf][m_off * 32], 32);
    v16h bf0 = load_frag_b(&sB[buf][n_off * 32], 32);
    v16h bf1 = load_frag_b(&sB[buf][(n_off + 16) * 32], 32);
    c0 = wmma_f32_f16(af, bf0, c0);
    c1 = wmma_f32_f16(af, bf1, c1);

    __syncthreads();                                          // done reading buf
    if (kt + 2 < nk && tid < 32) {
      const int k2 = (kt + 2) << 5;
      tdm_load_tile_f16(A  + (size_t)mBase * K + k2, lds_offset_of(&sA[buf][0]), 32, 64, K);
      tdm_load_tile_f16(BT + (size_t)nBase * K + k2, lds_offset_of(&sB[buf][0]), 32, 64, K);
    }
  }

  const int c   = lane & 15;
  const int hi8 = (lane >> 4) << 3;
  const float bv0 = bias[nBase + n_off + c];
  const float bv1 = bias[nBase + n_off + 16 + c];
#pragma unroll
  for (int r = 0; r < 8; ++r) {
    const int m  = mBase + m_off + hi8 + r;
    const int n0 = nBase + n_off + c;
    const int n1 = n0 + 16;
    float x0 = c0[r] + bv0;
    float x1 = c1[r] + bv1;
    if (RELU) { x0 = fmaxf(x0, 0.f); x1 = fmaxf(x1, 0.f); }
    if (OUT32)  { C32[(size_t)m * N + n0] = x0;            C32[(size_t)m * N + n1] = x1; }
    if (OUT16)  { C16[(size_t)m * N + n0] = (_Float16)x0;  C16[(size_t)m * N + n1] = (_Float16)x1; }
    if (OUT16T) { C16T[(size_t)n0 * M + m] = (_Float16)x0; C16T[(size_t)n1 * M + m] = (_Float16)x1; }
  }
}

// ---------------------------------------------------------------------------
// Flash attention with fused geometric bias (one wave per b,h,16-row tile).
// ---------------------------------------------------------------------------
__global__ __launch_bounds__(32)
void attn_flash_wmma(const _Float16* __restrict__ q16,   // [ROWS x DM]
                     const _Float16* __restrict__ k16,   // [ROWS x DM]
                     const _Float16* __restrict__ vT16,  // [DM x ROWS]
                     const float*    __restrict__ coords,// [ROWS x 3]
                     const float*    __restrict__ alpha, // [NLAYER]
                     int layer,
                     float*          __restrict__ attn_out) // [ROWS x DM]
{
  __shared__ __align__(16) _Float16 pbuf[16 * 32];

  const int lane = threadIdx.x & 31;
  const int c    = lane & 15;
  const int hi8  = (lane >> 4) << 3;
  const int m0   = blockIdx.x * 16;
  const int h    = blockIdx.y;
  const int b    = blockIdx.z;
  const int gm   = b * SEQ;

  const float scale  = 0.17677669529663687f;   // 1/sqrt(32)
  const float alphaL = alpha[layer];

  const v16h qa = load_frag_a(q16 + (size_t)(gm + m0) * DM + h * DKH, DM);

  float rcx[8], rcy[8], rcz[8];
#pragma unroll
  for (int r = 0; r < 8; ++r) {
    const float* pc = coords + (size_t)(gm + m0 + hi8 + r) * 3;
    rcx[r] = pc[0]; rcy[r] = pc[1]; rcz[r] = pc[2];
  }

  v8f o0 = {0.f,0.f,0.f,0.f,0.f,0.f,0.f,0.f};
  v8f o1 = {0.f,0.f,0.f,0.f,0.f,0.f,0.f,0.f};
  float mrun[8], lrun[8];
#pragma unroll
  for (int r = 0; r < 8; ++r) { mrun[r] = -3.0e38f; lrun[r] = 0.f; }

  for (int j = 0; j < SEQ; j += 32) {
    const v16h kb0 = load_frag_b(k16 + (size_t)(gm + j)      * DM + h * DKH, DM);
    const v16h kb1 = load_frag_b(k16 + (size_t)(gm + j + 16) * DM + h * DKH, DM);
    const v8f z = {0.f,0.f,0.f,0.f,0.f,0.f,0.f,0.f};
    v8f s0 = wmma_f32_f16(qa, kb0, z);
    v8f s1 = wmma_f32_f16(qa, kb1, z);

    const float* pn0 = coords + (size_t)(gm + j + c) * 3;
    const float* pn1 = coords + (size_t)(gm + j + 16 + c) * 3;
    const float n0x = pn0[0], n0y = pn0[1], n0z = pn0[2];
    const float n1x = pn1[0], n1y = pn1[1], n1z = pn1[2];

    float p0[8], p1[8];
#pragma unroll
    for (int r = 0; r < 8; ++r) {
      const float g0 = rcx[r]*n0x + rcy[r]*n0y + rcz[r]*n0z;
      const float g1 = rcx[r]*n1x + rcy[r]*n1y + rcz[r]*n1z;
      const float t0 = s0[r]*scale + alphaL*g0;
      const float t1 = s1[r]*scale + alphaL*g1;
      const float mx   = dpp_bcast_max16(fmaxf(t0, t1));     // per-row-half max
      const float mnew = fmaxf(mrun[r], mx);
      const float f    = __expf(mrun[r] - mnew);
      const float e0   = __expf(t0 - mnew);
      const float e1   = __expf(t1 - mnew);
      const float rs   = dpp_bcast_sum16(e0 + e1);           // per-row-half sum
      lrun[r] = lrun[r] * f + rs;
      mrun[r] = mnew;
      o0[r] *= f; o1[r] *= f;
      p0[r] = e0; p1[r] = e1;
    }

    // C-layout -> A-layout through LDS for the PV contraction
#pragma unroll
    for (int r = 0; r < 8; ++r) {
      pbuf[(hi8 + r) * 32 + c]      = (_Float16)p0[r];
      pbuf[(hi8 + r) * 32 + 16 + c] = (_Float16)p1[r];
    }
    __syncthreads();
    const v16h pa  = load_frag_a(pbuf, 32);
    const v16h vb0 = load_frag_b(vT16 + (size_t)(h * DKH)      * ROWS + gm + j, ROWS);
    const v16h vb1 = load_frag_b(vT16 + (size_t)(h * DKH + 16) * ROWS + gm + j, ROWS);
    o0 = wmma_f32_f16(pa, vb0, o0);
    o1 = wmma_f32_f16(pa, vb1, o1);
    __syncthreads();
  }

#pragma unroll
  for (int r = 0; r < 8; ++r) {
    const float inv = 1.0f / lrun[r];
    const int m = gm + m0 + hi8 + r;
    attn_out[(size_t)m * DM + h * DKH + c]      = o0[r] * inv;
    attn_out[(size_t)m * DM + h * DKH + 16 + c] = o1[r] * inv;
  }
}

// ---------------------------------------------------------------------------
// x = LayerNorm(xin + res) * g + b ; writes f32 and f16 copies.
// ---------------------------------------------------------------------------
__global__ __launch_bounds__(256)
void residual_layernorm(const float* __restrict__ xin, const float* __restrict__ res,
                        const float* __restrict__ g,   const float* __restrict__ bt,
                        float* __restrict__ xout, _Float16* __restrict__ xout16)
{
  __shared__ float red[8];
  const int row = blockIdx.x;
  const int t   = threadIdx.x;
  const float v = xin[(size_t)row * DM + t] + res[(size_t)row * DM + t];

  float s = v;
#pragma unroll
  for (int off = 1; off < 32; off <<= 1) s += __shfl_xor(s, off, 32);
  if ((t & 31) == 0) red[t >> 5] = s;
  __syncthreads();
  float mu = 0.f;
#pragma unroll
  for (int i = 0; i < 8; ++i) mu += red[i];
  mu *= (1.f / DM);
  __syncthreads();

  const float d = v - mu;
  float sq = d * d;
#pragma unroll
  for (int off = 1; off < 32; off <<= 1) sq += __shfl_xor(sq, off, 32);
  if ((t & 31) == 0) red[t >> 5] = sq;
  __syncthreads();
  float var = 0.f;
#pragma unroll
  for (int i = 0; i < 8; ++i) var += red[i];
  var *= (1.f / DM);

  const float y = d * rsqrtf(var + LN_EPS) * g[t] + bt[t];
  xout  [(size_t)row * DM + t] = y;
  xout16[(size_t)row * DM + t] = (_Float16)y;
}

// ---------------------------------------------------------------------------
// Conversions
// ---------------------------------------------------------------------------
__global__ void convert_x(const float* __restrict__ x, float* __restrict__ xf,
                          _Float16* __restrict__ x16, int n) {
  int i = blockIdx.x * 256 + threadIdx.x;
  if (i < n) { const float v = x[i]; xf[i] = v; x16[i] = (_Float16)v; }
}

// wT[n*K + k] = (f16) w[k*N + n]
__global__ void convert_wT(const float* __restrict__ w, _Float16* __restrict__ wT,
                           int K, int N) {
  int i = blockIdx.x * 256 + threadIdx.x;
  if (i < K * N) {
    const int k = i / N, n = i % N;
    wT[(size_t)n * K + k] = (_Float16)w[i];
  }
}

// ---------------------------------------------------------------------------
// Host-side orchestration
// ---------------------------------------------------------------------------
extern "C" void kernel_launch(void* const* d_in, const int* in_sizes, int n_in,
                              void* d_out, int out_size, void* d_ws, size_t ws_size,
                              hipStream_t stream) {
  (void)in_sizes; (void)n_in; (void)out_size; (void)ws_size;
  const float* x      = (const float*)d_in[0];
  const float* coords = (const float*)d_in[1];
  const float* wq     = (const float*)d_in[2];
  const float* bq     = (const float*)d_in[3];
  const float* wk     = (const float*)d_in[4];
  const float* bk     = (const float*)d_in[5];
  const float* wv     = (const float*)d_in[6];
  const float* bv     = (const float*)d_in[7];
  const float* alpha  = (const float*)d_in[8];
  const float* w1     = (const float*)d_in[9];
  const float* b1     = (const float*)d_in[10];
  const float* w2     = (const float*)d_in[11];
  const float* b2     = (const float*)d_in[12];
  const float* ln1g   = (const float*)d_in[13];
  const float* ln1b   = (const float*)d_in[14];
  const float* ln2g   = (const float*)d_in[15];
  const float* ln2b   = (const float*)d_in[16];
  float* out = (float*)d_out;

  char* ws = (char*)d_ws;
  size_t off = 0;
  auto carve = [&](size_t bytes) -> void* {
    void* p = ws + off;
    off += (bytes + 255) & ~(size_t)255;
    return p;
  };
  float*    xf   = (float*)   carve((size_t)ROWS * DM * 4);
  _Float16* x16  = (_Float16*)carve((size_t)ROWS * DM * 2);
  _Float16* q16  = (_Float16*)carve((size_t)ROWS * DM * 2);
  _Float16* k16  = (_Float16*)carve((size_t)ROWS * DM * 2);
  _Float16* vT16 = (_Float16*)carve((size_t)ROWS * DM * 2);
  float*    attn = (float*)   carve((size_t)ROWS * DM * 4);
  _Float16* h116 = (_Float16*)carve((size_t)ROWS * DFF * 2);
  float*    ff   = (float*)   carve((size_t)ROWS * DM * 4);
  _Float16* wqT  = (_Float16*)carve((size_t)NLAYER * DM * DM * 2);
  _Float16* wkT  = (_Float16*)carve((size_t)NLAYER * DM * DM * 2);
  _Float16* wvT  = (_Float16*)carve((size_t)NLAYER * DM * DM * 2);
  _Float16* w1T  = (_Float16*)carve((size_t)NLAYER * DM * DFF * 2);
  _Float16* w2T  = (_Float16*)carve((size_t)NLAYER * DFF * DM * 2);

  const int gDD = (DM * DM + 255) / 256;
  const int gDF = (DM * DFF + 255) / 256;
  for (int l = 0; l < NLAYER; ++l) {
    convert_wT<<<gDD, 256, 0, stream>>>(wq + (size_t)l * DM * DM,  wqT + (size_t)l * DM * DM,  DM, DM);
    convert_wT<<<gDD, 256, 0, stream>>>(wk + (size_t)l * DM * DM,  wkT + (size_t)l * DM * DM,  DM, DM);
    convert_wT<<<gDD, 256, 0, stream>>>(wv + (size_t)l * DM * DM,  wvT + (size_t)l * DM * DM,  DM, DM);
    convert_wT<<<gDF, 256, 0, stream>>>(w1 + (size_t)l * DM * DFF, w1T + (size_t)l * DM * DFF, DM,  DFF);
    convert_wT<<<gDF, 256, 0, stream>>>(w2 + (size_t)l * DFF * DM, w2T + (size_t)l * DFF * DM, DFF, DM);
  }
  convert_x<<<(ROWS * DM + 255) / 256, 256, 0, stream>>>(x, xf, x16, ROWS * DM);

  const dim3 gQKV(ROWS / 64, DM / 64);
  const dim3 gF1 (ROWS / 64, DFF / 64);
  const dim3 gF2 (ROWS / 64, DM / 64);
  const dim3 gAtt(SEQ / 16, NHEAD, BATCH);

  for (int l = 0; l < NLAYER; ++l) {
    gemm_f16_wmma<false, false, true,  false><<<gQKV, 256, 0, stream>>>(
        x16, wqT + (size_t)l * DM * DM, bq + (size_t)l * DM,
        nullptr, q16, nullptr, ROWS, DM, DM);
    gemm_f16_wmma<false, false, true,  false><<<gQKV, 256, 0, stream>>>(
        x16, wkT + (size_t)l * DM * DM, bk + (size_t)l * DM,
        nullptr, k16, nullptr, ROWS, DM, DM);
    gemm_f16_wmma<false, false, false, true ><<<gQKV, 256, 0, stream>>>(
        x16, wvT + (size_t)l * DM * DM, bv + (size_t)l * DM,
        nullptr, nullptr, vT16, ROWS, DM, DM);

    attn_flash_wmma<<<gAtt, 32, 0, stream>>>(q16, k16, vT16, coords, alpha, l, attn);

    residual_layernorm<<<ROWS, 256, 0, stream>>>(
        xf, attn, ln1g + (size_t)l * DM, ln1b + (size_t)l * DM, xf, x16);

    gemm_f16_wmma<true,  false, true,  false><<<gF1, 256, 0, stream>>>(
        x16, w1T + (size_t)l * DM * DFF, b1 + (size_t)l * DFF,
        nullptr, h116, nullptr, ROWS, DFF, DM);
    gemm_f16_wmma<false, true,  false, false><<<gF2, 256, 0, stream>>>(
        h116, w2T + (size_t)l * DFF * DM, b2 + (size_t)l * DM,
        ff, nullptr, nullptr, ROWS, DM, DFF);

    float* xdst = (l == NLAYER - 1) ? out : xf;
    residual_layernorm<<<ROWS, 256, 0, stream>>>(
        xf, ff, ln2g + (size_t)l * DM, ln2b + (size_t)l * DM, xdst, x16);
  }
}
